// Non_Local_Module_14877766713778
// MI455X (gfx1250) — matrix operand order repaired and checked by reading, compiled, not run on hardware
//
#include <hip/hip_runtime.h>
#include <hip/hip_bf16.h>

typedef __attribute__((ext_vector_type(16))) _Float16 v16h;
typedef __attribute__((ext_vector_type(8)))  float    v8f;
typedef __attribute__((ext_vector_type(4)))  _Float16 h16x4;
typedef __attribute__((ext_vector_type(4)))  float    f32x4;
typedef __attribute__((ext_vector_type(4)))  unsigned int u32x4;
typedef __attribute__((ext_vector_type(8)))  unsigned int u32x8;

#define C1 256
#define CI 128
#define HW 4096   // 64*64
#define MP 1024   // pooled spatial (32*32)

__device__ __forceinline__ v8f vzero8() {
  v8f z = {0.f,0.f,0.f,0.f,0.f,0.f,0.f,0.f};
  return z;
}

__device__ __forceinline__ v8f wmma_f16(v16h a, v16h b, v8f c) {
  return __builtin_amdgcn_wmma_f32_16x16x32_f16(
      false, a, false, b, (short)0, c, false, false);
}

// A fragment: 16x32 (MxK) f16.  base -> (row0,k0) of tile, row-major, K contiguous.
// Row m = lane&15; lanes 0-15 hold K {0..7,16..23}, lanes 16-31 K {8..15,24..31};
// VGPR i packs K = (i>=4?16:0)+kh*8+(i%4)*2 +{0,1}.   (ISA 7.12.2)
__device__ __forceinline__ v16h load_a_frag(const _Float16* base, int stride, int lane) {
  const int m  = lane & 15;
  const int kh = lane >> 4;
  const _Float16* p = base + m * stride;
  v16h a;
#pragma unroll
  for (int i = 0; i < 8; ++i) {
    const int k = ((i >> 2) << 4) + (kh << 3) + ((i & 3) << 1);
    a[2*i]   = p[k];
    a[2*i+1] = p[k+1];
  }
  return a;
}

// B fragment: 32x16 (KxN) f16. LDS stored transposed [col][K] (K contiguous).
// Col n = lane&15; lanes 0-15 hold K {0..7,16..23}, lanes 16-31 K {8..15,24..31} (same as A).
__device__ __forceinline__ v16h load_b_frag(const _Float16* base, int stride, int lane) {
  const int n  = lane & 15;
  const int kh = lane >> 4;
  const _Float16* p = base + n * stride + (kh << 3);
  v16h b;
#pragma unroll
  for (int i = 0; i < 8; ++i) {
    b[2*i]   = p[((i >> 2) << 4) + ((i & 3) << 1)];
    b[2*i+1] = p[((i >> 2) << 4) + ((i & 3) << 1) + 1];
  }
  return b;
}

// ---------------------------------------------------------------------------
// Kernel 1: 1x1-conv projections.  out[b][m][n] = sum_k w[m][k] x[b][k][n] + bias
// M=128, K=256, N=4096.  Block computes 128x128 tile; proj = blockIdx.z.
// ---------------------------------------------------------------------------
__global__ __launch_bounds__(256)
void proj_kernel(const float* __restrict__ x,
                 const float* __restrict__ tw, const float* __restrict__ tb,
                 const float* __restrict__ pw, const float* __restrict__ pb,
                 const float* __restrict__ gw, const float* __restrict__ gb,
                 _Float16* __restrict__ theta_o,
                 _Float16* __restrict__ phi_o,
                 _Float16* __restrict__ g_o) {
  const int b     = blockIdx.y;
  const int proj  = blockIdx.z;
  const int ntile = blockIdx.x * 128;
  const float* w    = (proj == 0) ? tw : (proj == 1) ? pw : gw;
  const float* bias = (proj == 0) ? tb : (proj == 1) ? pb : gb;
  _Float16* outp    = (proj == 0) ? theta_o : (proj == 1) ? phi_o : g_o;

  __shared__ _Float16 wl[CI * 34];   // [m][k] stride 34
  __shared__ _Float16 xl[128 * 34];  // [n][k] stride 34 (transposed: K contiguous)
  __shared__ float    bl[CI];

  const int tid  = threadIdx.x;
  const int lane = tid & 31;
  const int wv   = tid >> 5;
  const int mh   = lane >> 4;
  const int nn   = lane & 15;

  if (tid < CI) bl[tid] = bias[tid];

  v8f acc[8];
#pragma unroll
  for (int s = 0; s < 8; ++s) acc[s] = vzero8();

  for (int kt = 0; kt < C1; kt += 32) {
    __syncthreads();
    // weights: 16B vector loads, K contiguous in LDS
#pragma unroll
    for (int e = tid; e < CI * 8; e += 256) {
      const int k4 = (e & 7) * 4, m = e >> 3;
      const f32x4 wv4 = *(const f32x4*)(w + m * C1 + kt + k4);
#pragma unroll
      for (int j = 0; j < 4; ++j) wl[m * 34 + k4 + j] = (_Float16)wv4[j];
    }
    // x tile: 16B vector loads coalesced in n, transposed scatter to LDS
#pragma unroll
    for (int e = tid; e < 32 * 32; e += 256) {
      const int n4 = (e & 31) * 4, k = e >> 5;
      const f32x4 xv = *(const f32x4*)(x + ((size_t)b * C1 + kt + k) * HW + ntile + n4);
#pragma unroll
      for (int j = 0; j < 4; ++j) xl[(n4 + j) * 34 + k] = (_Float16)xv[j];
    }
    __syncthreads();

    const v16h a = load_a_frag(wl + wv * 16 * 34, 34, lane);
#pragma unroll
    for (int s = 0; s < 8; ++s) {
      const v16h bb = load_b_frag(xl + s * 16 * 34, 34, lane);
      acc[s] = wmma_f16(a, bb, acc[s]);
    }
  }

  const int mrow0 = wv * 16 + mh * 8;
#pragma unroll
  for (int s = 0; s < 8; ++s) {
#pragma unroll
    for (int i = 0; i < 8; ++i) {
      const float v = acc[s][i] + bl[mrow0 + i];
      outp[((size_t)b * CI + mrow0 + i) * HW + ntile + s * 16 + nn] = (_Float16)v;
    }
  }
}

// ---------------------------------------------------------------------------
// Kernel 2: 2x2 max pool (64x64 -> 32x32), f16 in/out.
// ---------------------------------------------------------------------------
__global__ __launch_bounds__(256)
void pool_kernel(const _Float16* __restrict__ in, _Float16* __restrict__ out,
                 int total) {
  const int idx = blockIdx.x * 256 + threadIdx.x;
  if (idx >= total) return;
  const int m  = idx & (MP - 1);
  const int bc = idx >> 10;                 // b*128 + c
  const int i = m >> 5, j = m & 31;
  const _Float16* p = in + (size_t)bc * HW + (i * 2) * 64 + j * 2;
  const float v0 = (float)p[0],  v1 = (float)p[1];
  const float v2 = (float)p[64], v3 = (float)p[65];
  out[idx] = (_Float16)fmaxf(fmaxf(v0, v1), fmaxf(v2, v3));
}

// ---------------------------------------------------------------------------
// Kernel 3: flash attention.  Per block: 128 queries x 1024 keys, Ci=128.
// S = theta^T phi, online softmax, O += P g^T.  y out as [b][n][c] f16.
// g tiles are DMA'd into LDS by the Tensor Data Mover (pad 1dw/32dw -> stride 66).
// ---------------------------------------------------------------------------
__global__ __launch_bounds__(256)
void attn_kernel(const _Float16* __restrict__ theta,  // [b][c][n]
                 const _Float16* __restrict__ phi,    // [b][c][m]
                 const _Float16* __restrict__ gmat,   // [b][c][m]
                 _Float16* __restrict__ y) {          // [b][n][c]
  const int b     = blockIdx.y;
  const int qbase = blockIdx.x * 128;

  // Unioned LDS: theta staging (33,280 B) overlaps phl+gl; steady state 50,432 B.
  __shared__ char smem[50432];
  _Float16* phl = (_Float16*)smem;            // [key64][c] stride 130 -> 16,640 B
  _Float16* gl  = (_Float16*)(smem + 16640);  // [c128][key] stride 66 -> 16,896 B (TDM)
  _Float16* pl  = (_Float16*)(smem + 33536);  // 8 waves x [q16][key64] stride 66
  _Float16* thl = (_Float16*)smem;            // [q128][c] stride 130 (setup only)

  const int tid  = threadIdx.x;
  const int lane = tid & 31;
  const int wv   = tid >> 5;
  const int mh   = lane >> 4;
  const int nn   = lane & 15;

  // Stage theta transposed [q][c]: 8B global loads coalesced in q.
#pragma unroll
  for (int e = tid; e < 128 * 32; e += 256) {
    const int q4 = (e & 31) * 4, c = e >> 5;
    const h16x4 tv = *(const h16x4*)(theta + ((size_t)b * CI + c) * HW + qbase + q4);
#pragma unroll
    for (int j = 0; j < 4; ++j) thl[(q4 + j) * 130 + c] = tv[j];
  }
  __syncthreads();

  v16h thA[4];
#pragma unroll
  for (int ks = 0; ks < 4; ++ks)
    thA[ks] = load_a_frag(thl + (wv * 16) * 130 + ks * 32, 130, lane);
  __syncthreads();  // frags extracted; thl region may now be reused

  // TDM descriptor for the g tile: tensor [c=128][m=1024] f16, tile 64(m) x 128(c),
  // row = 64 elems = 32 dwords; pad 1 dword after every 32 -> LDS stride 66 elems.
  u32x4 tg0;
  tg0[0] = 1u;                                   // count=1, user descriptor
  tg0[1] = (unsigned)(size_t)(void*)gl;          // lds_addr (low 32 bits of generic)
  tg0[2] = 0u; tg0[3] = 0u;                      // global addr filled per tile
  u32x8 tg1;
  tg1[0] = (1u << 16) | (1u << 20) | (4u << 22); // data_size=2B | pad_en | interval=32dw (amount=1dw)
  tg1[1] = (MP & 0xFFFFu) << 16;                 // tensor_dim0 = 1024 (bits 79:48)
  tg1[2] = (CI & 0xFFFFu) << 16;                 // tensor_dim1 = 128 (bits 111:80)
  tg1[3] = 64u << 16;                            // tile_dim0 = 64 (bits 127:112)
  tg1[4] = 128u;                                 // tile_dim1 = 128, tile_dim2 = 0
  tg1[5] = MP;                                   // tensor_dim0_stride = 1024 (low 32)
  tg1[6] = 0u;
  tg1[7] = 0u;

  v8f o[8];
#pragma unroll
  for (int f = 0; f < 8; ++f) o[f] = vzero8();
  float mrun[8], lrun[8];
#pragma unroll
  for (int i = 0; i < 8; ++i) { mrun[i] = -1.0e30f; lrun[i] = 0.0f; }

  _Float16* pw = pl + wv * 16 * 66;

  for (int kb = 0; kb < 16; ++kb) {
    __syncthreads();
    if (wv == 0) {
      // kick the TDM first so the DMA overlaps the VALU phi staging below
      const unsigned long long ga =
          (unsigned long long)(size_t)(gmat + (size_t)b * CI * MP + kb * 64);
      u32x4 d0 = tg0;
      d0[2] = (unsigned)ga;
      d0[3] = (unsigned)(ga >> 32) | (2u << 30);   // global_addr[56:32] | type=2
      asm volatile("tensor_load_to_lds %0, %1" :: "s"(d0), "s"(tg1) : "memory");
    }
    // phi [key][c]: 8B global loads coalesced in key, transposed scatter
#pragma unroll
    for (int e = tid; e < 64 * 32; e += 256) {
      const int key4 = (e & 15) * 4, c = e >> 4;
      const h16x4 pv = *(const h16x4*)(phi + ((size_t)b * CI + c) * MP + kb * 64 + key4);
#pragma unroll
      for (int j = 0; j < 4; ++j) phl[(key4 + j) * 130 + c] = pv[j];
    }
    if (wv == 0) __builtin_amdgcn_s_wait_tensorcnt(0);
    __syncthreads();

    // S = theta^T phi : 4 key-subtiles x 4 K-steps of 32 channels
    v8f s[4];
#pragma unroll
    for (int t = 0; t < 4; ++t) {
      v8f st = vzero8();
#pragma unroll
      for (int ks = 0; ks < 4; ++ks) {
        const v16h bb = load_b_frag(phl + t * 16 * 130 + ks * 32, 130, lane);
        st = wmma_f16(thA[ks], bb, st);
      }
      s[t] = st;
    }

    // online softmax: row stats via 16-lane butterfly within each half
    float tmax[8];
#pragma unroll
    for (int i = 0; i < 8; ++i)
      tmax[i] = fmaxf(fmaxf(s[0][i], s[1][i]), fmaxf(s[2][i], s[3][i]));
#pragma unroll
    for (int msk = 1; msk < 16; msk <<= 1)
#pragma unroll
      for (int i = 0; i < 8; ++i)
        tmax[i] = fmaxf(tmax[i], __shfl_xor(tmax[i], msk, 32));

    float sc[8];
#pragma unroll
    for (int i = 0; i < 8; ++i) {
      const float mnew = fmaxf(mrun[i], tmax[i]);
      sc[i]   = __expf(mrun[i] - mnew);
      mrun[i] = mnew;
    }

    float rs[8];
#pragma unroll
    for (int i = 0; i < 8; ++i) rs[i] = 0.0f;
#pragma unroll
    for (int t = 0; t < 4; ++t)
#pragma unroll
      for (int i = 0; i < 8; ++i) {
        const float pv = __expf(s[t][i] - mrun[i]);
        s[t][i] = pv;
        rs[i] += pv;
      }
#pragma unroll
    for (int msk = 1; msk < 16; msk <<= 1)
#pragma unroll
      for (int i = 0; i < 8; ++i)
        rs[i] += __shfl_xor(rs[i], msk, 32);
#pragma unroll
    for (int i = 0; i < 8; ++i) lrun[i] = lrun[i] * sc[i] + rs[i];
#pragma unroll
    for (int f = 0; f < 8; ++f)
#pragma unroll
      for (int i = 0; i < 8; ++i) o[f][i] *= sc[i];

    // P (D-layout) -> wave-private LDS [q][key], read back as A-frags
#pragma unroll
    for (int t = 0; t < 4; ++t)
#pragma unroll
      for (int i = 0; i < 8; ++i)
        pw[(mh * 8 + i) * 66 + t * 16 + nn] = (_Float16)s[t][i];

    // O += P g^T : 2 K-steps of 32 keys x 8 channel-subtiles
#pragma unroll
    for (int ks = 0; ks < 2; ++ks) {
      const v16h aP = load_a_frag(pw + ks * 32, 66, lane);
#pragma unroll
      for (int cf = 0; cf < 8; ++cf) {
        const v16h bb = load_b_frag(gl + cf * 16 * 66 + ks * 32, 66, lane);
        o[cf] = wmma_f16(aP, bb, o[cf]);
      }
    }
  }

  // finalize: O /= l, store y[b][n][c] f16
  float rl[8];
#pragma unroll
  for (int i = 0; i < 8; ++i) rl[i] = 1.0f / lrun[i];
#pragma unroll
  for (int cf = 0; cf < 8; ++cf)
#pragma unroll
    for (int i = 0; i < 8; ++i) {
      const float v = o[cf][i] * rl[i];
      const int q = qbase + wv * 16 + mh * 8 + i;
      y[((size_t)b * HW + q) * CI + cf * 16 + nn] = (_Float16)v;
    }
}

// ---------------------------------------------------------------------------
// Kernel 4: output 1x1 conv + BatchNorm.
// out[b][o][n] = (sum_c W[o][c] y[b][n][c] + Wb[o]) * inv[o] + (beta - mean*inv)[o]
// M=256, K=128, N=4096; block tile 256x64.
// ---------------------------------------------------------------------------
__global__ __launch_bounds__(256)
void out_kernel(const _Float16* __restrict__ y,   // [b][n][c]
                const float* __restrict__ Ww, const float* __restrict__ Wb,
                const float* __restrict__ bg, const float* __restrict__ bb,
                const float* __restrict__ bm, const float* __restrict__ bv,
                float* __restrict__ out) {
  const int b     = blockIdx.y;
  const int ntile = blockIdx.x * 64;

  __shared__ _Float16 wl[C1 * 34];   // per-K-slice W [o][k32] stride 34
  __shared__ _Float16 yl[64 * 132];  // [n][c] stride 132 (row bytes 264, 8-aligned)
  __shared__ float    scl[C1], sft[C1];

  const int tid  = threadIdx.x;
  const int lane = tid & 31;
  const int wv   = tid >> 5;
  const int mh   = lane >> 4;
  const int nn   = lane & 15;

  {
    const float iv = bg[tid] * rsqrtf(bv[tid] + 1e-5f);
    scl[tid] = iv;
    sft[tid] = bb[tid] - bm[tid] * iv + Wb[tid] * iv;
  }
  // y tile: straight 8B copies (layouts match), both sides 8-byte aligned
#pragma unroll
  for (int e = tid; e < 64 * 32; e += 256) {
    const int c4 = (e & 31) * 4, n = e >> 5;
    *(unsigned long long*)(yl + n * 132 + c4) =
        *(const unsigned long long*)(y + ((size_t)b * HW + ntile + n) * CI + c4);
  }

  v8f acc[2][4];
#pragma unroll
  for (int h = 0; h < 2; ++h)
#pragma unroll
    for (int ns = 0; ns < 4; ++ns) acc[h][ns] = vzero8();

  for (int ks = 0; ks < 4; ++ks) {
    __syncthreads();
#pragma unroll
    for (int e = tid; e < C1 * 8; e += 256) {
      const int k4 = (e & 7) * 4, oo = e >> 3;
      const f32x4 wv4 = *(const f32x4*)(Ww + oo * CI + ks * 32 + k4);
#pragma unroll
      for (int j = 0; j < 4; ++j) wl[oo * 34 + k4 + j] = (_Float16)wv4[j];
    }
    __syncthreads();
    const v16h a0 = load_a_frag(wl + (wv * 32) * 34, 34, lane);
    const v16h a1 = load_a_frag(wl + (wv * 32 + 16) * 34, 34, lane);
#pragma unroll
    for (int ns = 0; ns < 4; ++ns) {
      const v16h bf = load_b_frag(yl + ns * 16 * 132 + ks * 32, 132, lane);
      acc[0][ns] = wmma_f16(a0, bf, acc[0][ns]);
      acc[1][ns] = wmma_f16(a1, bf, acc[1][ns]);
    }
  }

#pragma unroll
  for (int h = 0; h < 2; ++h) {
    const int ob = wv * 32 + h * 16 + mh * 8;
#pragma unroll
    for (int ns = 0; ns < 4; ++ns)
#pragma unroll
      for (int i = 0; i < 8; ++i) {
        const int oo = ob + i;
        const float v = acc[h][ns][i] * scl[oo] + sft[oo];
        out[((size_t)b * C1 + oo) * HW + ntile + ns * 16 + nn] = v;
      }
  }
}

// ---------------------------------------------------------------------------
extern "C" void kernel_launch(void* const* d_in, const int* in_sizes, int n_in,
                              void* d_out, int out_size, void* d_ws, size_t ws_size,
                              hipStream_t stream) {
  (void)in_sizes; (void)n_in; (void)out_size; (void)ws_size;
  const float* x  = (const float*)d_in[0];
  const float* tw = (const float*)d_in[1];
  const float* tb = (const float*)d_in[2];
  const float* pw = (const float*)d_in[3];
  const float* pb = (const float*)d_in[4];
  const float* gw = (const float*)d_in[5];
  const float* gb = (const float*)d_in[6];
  const float* Ww = (const float*)d_in[7];
  const float* Wb = (const float*)d_in[8];
  const float* bg = (const float*)d_in[9];
  const float* bb = (const float*)d_in[10];
  const float* bm = (const float*)d_in[11];
  const float* bv = (const float*)d_in[12];

  char* ws = (char*)d_ws;
  // f16 workspace layout (bytes): theta 8MB | phi_full 8MB | g_full 8MB |
  // phi_pool 2MB | g_pool 2MB.  y reuses the (dead) phi_full region.
  _Float16* theta_ws = (_Float16*)(ws);
  _Float16* phi_full = (_Float16*)(ws + (8u << 20));
  _Float16* g_full   = (_Float16*)(ws + (16u << 20));
  _Float16* phi_pool = (_Float16*)(ws + (24u << 20));
  _Float16* g_pool   = (_Float16*)(ws + (26u << 20));
  _Float16* y_ws     = (_Float16*)(ws + (8u << 20));   // alias phi_full (dead)

  const dim3 blk(256);
  proj_kernel<<<dim3(HW / 128, 8, 3), blk, 0, stream>>>(
      x, tw, tb, pw, pb, gw, gb, theta_ws, phi_full, g_full);

  const int ptot = 8 * CI * MP;
  pool_kernel<<<(ptot + 255) / 256, blk, 0, stream>>>(phi_full, phi_pool, ptot);
  pool_kernel<<<(ptot + 255) / 256, blk, 0, stream>>>(g_full, g_pool, ptot);

  attn_kernel<<<dim3(HW / 128, 8), blk, 0, stream>>>(
      theta_ws, phi_pool, g_pool, y_ws);

  out_kernel<<<dim3(HW / 64, 8), blk, 0, stream>>>(
      y_ws, Ww, Wb, bg, bb, bm, bv, (float*)d_out);
}
